// MultiheadAttention_4595615007090
// MI455X (gfx1250) — matrix operand order repaired and checked by reading, not compiled, not run on hardware
//
#include <hip/hip_runtime.h>

// ---------------------------------------------------------------------------
// CDNA5 (gfx1250) multi-head attention, bf16 WMMA pipeline, fp32 accumulate.
//   D_MODEL=512, NHEAD=4, HEAD_DIM=128, B=2, S=4096  ->  M = B*S = 8192
// All GEMM operands pre-converted to bf16 (one memory-bound pass, ~3us at
// 23.3 TB/s) so GEMM inner loops are pure b128-load + v_wmma.
// K/V staging in attention uses GLOBAL_LOAD_ASYNC_TO_LDS_B128 (ASYNCcnt).
// ---------------------------------------------------------------------------

typedef __attribute__((ext_vector_type(16))) __bf16       v16bf;
typedef __attribute__((ext_vector_type(8)))  float        v8f;
typedef __attribute__((ext_vector_type(4)))  unsigned int u32x4;
typedef __attribute__((ext_vector_type(2)))  unsigned int u32x2;
typedef __attribute__((ext_vector_type(4)))  float        f32x4;
typedef __attribute__((ext_vector_type(4)))  int          i32x4;

union BF16x16 { v16bf v; __bf16 e[16]; unsigned short s[16]; u32x4 q[2]; };
union BF16x4  { __bf16 e[4]; u32x2 d; };
union F32x8   { v8f   v; float f[8]; };

__device__ __forceinline__ unsigned short f2bf(float x) {
  union { __bf16 h; unsigned short s; } u;
  u.h = (__bf16)x;            // native v_cvt if the target has it
  return u.s;
}

__device__ __forceinline__ v8f wmma_bf16(v16bf a, v16bf b, v8f c) {
  // v_wmma_f32_16x16x32_bf16: D = A(16x32) * B(32x16) + C(16x16 f32)
  return __builtin_amdgcn_wmma_f32_16x16x32_bf16(
      /*neg_a=*/false, a, /*neg_b=*/false, b,
      /*c_mod=*/(short)0, c, /*reuse_a=*/false, /*reuse_b=*/false);
}

// gfx1250 async global->LDS copy path (ASYNCcnt-tracked).
// Probe-confirmed present in this toolchain; expected signature:
//   (v4i addrspace(1)* src, v4i addrspace(3)* dst, imm offset, imm cpol)
#if __has_builtin(__builtin_amdgcn_global_load_async_to_lds_b128) && \
    __has_builtin(__builtin_amdgcn_s_wait_asynccnt)
#define USE_ASYNC_LDS 1
#else
#define USE_ASYNC_LDS 0
#endif

#define AS1 __attribute__((address_space(1)))
#define AS3 __attribute__((address_space(3)))

// ---------------------------------------------------------------------------
// fp32 -> bf16, 4 elements per thread (vector load/store)
// ---------------------------------------------------------------------------
__global__ void mha_cvt_bf16x4(const float* __restrict__ src,
                               unsigned short* __restrict__ dst, int n4) {
  int i = blockIdx.x * blockDim.x + threadIdx.x;
  if (i < n4) {
    f32x4 f = ((const f32x4*)src)[i];
    BF16x4 o;
    o.e[0] = (__bf16)f.x; o.e[1] = (__bf16)f.y;
    o.e[2] = (__bf16)f.z; o.e[3] = (__bf16)f.w;
    ((u32x2*)dst)[i] = o.d;
  }
}

// ---------------------------------------------------------------------------
// Projection GEMM:  Out_bf16[M,512] = bf16( (Xb @ W^T + bias) * scale )
// Xb, W both bf16; W row-major [n,k] (torch Linear), so B[k,n] = W[n,k].
// One wave computes a 64x32 output tile (4 x 2 WMMA tiles), K swept by 32.
// ---------------------------------------------------------------------------
__global__ __launch_bounds__(128)
void mha_proj_gemm(const unsigned short* __restrict__ Xb,
                   const unsigned short* __restrict__ W,
                   const float* __restrict__ bias,
                   unsigned short* __restrict__ Out,
                   float scale) {
  const int K = 512, N = 512;
  const int wid  = (blockIdx.x * 128 + threadIdx.x) >> 5;
  const int lane = threadIdx.x & 31;
  const int half = lane >> 4, l16 = lane & 15;
  const int mt = wid >> 4;                // 16 n-tiles of 32
  const int nt = wid & 15;
  const int m0 = mt * 64, n0 = nt * 32;

  F32x8 acc[4][2];
#pragma unroll
  for (int i = 0; i < 4; i++)
#pragma unroll
    for (int j = 0; j < 2; j++)
#pragma unroll
      for (int r = 0; r < 8; r++) acc[i][j].f[r] = 0.0f;

  for (int k0 = 0; k0 < K; k0 += 32) {
    // A: 16x32 bf16; lane<16: row=l16, K elems k0..k0+15
    //    lane>=16: same rows, K base +16
    BF16x16 A[4];
#pragma unroll
    for (int i = 0; i < 4; i++) {
      const unsigned short* p =
          Xb + (size_t)(m0 + 16 * i + l16) * K + k0 + half * 16;
      A[i].q[0] = *(const u32x4*)p;
      A[i].q[1] = *(const u32x4*)(p + 8);
    }
    // B: 32x16 bf16; lane column n = n0+16j+l16, K = k0+half*16 .. +15
    BF16x16 Bf[2];
#pragma unroll
    for (int j = 0; j < 2; j++) {
      const unsigned short* p =
          W + (size_t)(n0 + 16 * j + l16) * K + k0 + half * 16;
      Bf[j].q[0] = *(const u32x4*)p;
      Bf[j].q[1] = *(const u32x4*)(p + 8);
    }
#pragma unroll
    for (int i = 0; i < 4; i++)
#pragma unroll
      for (int j = 0; j < 2; j++)
        acc[i][j].v = wmma_bf16(A[i].v, Bf[j].v, acc[i][j].v);
  }
  // C layout: VGPR r -> M=r (lanes 0-15) / M=r+8 (lanes 16-31), N=l16
#pragma unroll
  for (int i = 0; i < 4; i++)
#pragma unroll
    for (int j = 0; j < 2; j++) {
      const int col = n0 + 16 * j + l16;
      const float bcol = bias[col];
#pragma unroll
      for (int r = 0; r < 8; r++) {
        const int row = m0 + 16 * i + r + half * 8;
        Out[(size_t)row * N + col] = f2bf((acc[i][j].f[r] + bcol) * scale);
      }
    }
}

// ---------------------------------------------------------------------------
// Output GEMM: Out_f32[M,512] = Cx_bf16[M,512] @ Wo^T + bo
// ---------------------------------------------------------------------------
__global__ __launch_bounds__(128)
void mha_out_gemm(const unsigned short* __restrict__ Xb,
                  const unsigned short* __restrict__ W,
                  const float* __restrict__ bias,
                  float* __restrict__ Out) {
  const int K = 512, N = 512;
  const int wid  = (blockIdx.x * 128 + threadIdx.x) >> 5;
  const int lane = threadIdx.x & 31;
  const int half = lane >> 4, l16 = lane & 15;
  const int mt = wid >> 4;
  const int nt = wid & 15;
  const int m0 = mt * 64, n0 = nt * 32;

  F32x8 acc[4][2];
#pragma unroll
  for (int i = 0; i < 4; i++)
#pragma unroll
    for (int j = 0; j < 2; j++)
#pragma unroll
      for (int r = 0; r < 8; r++) acc[i][j].f[r] = 0.0f;

  for (int k0 = 0; k0 < K; k0 += 32) {
    BF16x16 A[4];
#pragma unroll
    for (int i = 0; i < 4; i++) {
      const unsigned short* p =
          Xb + (size_t)(m0 + 16 * i + l16) * K + k0 + half * 16;
      A[i].q[0] = *(const u32x4*)p;
      A[i].q[1] = *(const u32x4*)(p + 8);
    }
    BF16x16 Bf[2];
#pragma unroll
    for (int j = 0; j < 2; j++) {
      const unsigned short* p =
          W + (size_t)(n0 + 16 * j + l16) * K + k0 + half * 16;
      Bf[j].q[0] = *(const u32x4*)p;
      Bf[j].q[1] = *(const u32x4*)(p + 8);
    }
#pragma unroll
    for (int i = 0; i < 4; i++)
#pragma unroll
      for (int j = 0; j < 2; j++)
        acc[i][j].v = wmma_bf16(A[i].v, Bf[j].v, acc[i][j].v);
  }
#pragma unroll
  for (int i = 0; i < 4; i++)
#pragma unroll
    for (int j = 0; j < 2; j++) {
      const int col = n0 + 16 * j + l16;
      const float bcol = bias[col];
#pragma unroll
      for (int r = 0; r < 8; r++) {
        const int row = m0 + 16 * i + r + half * 8;
        Out[(size_t)row * N + col] = acc[i][j].f[r] + bcol;
      }
    }
}

// ---------------------------------------------------------------------------
// Flash attention: block = 4 waves = 64 queries, one (b,h) per block.
// K/V key-blocks of 32 staged in LDS via async-to-LDS (ASYNCcnt).
// Each wave: 16 query rows, online softmax, O accum 16x128 f32.
// ---------------------------------------------------------------------------
__global__ __launch_bounds__(128)
void mha_attention(const unsigned short* __restrict__ Q,
                   const unsigned short* __restrict__ Kg,
                   const unsigned short* __restrict__ Vg,
                   unsigned short* __restrict__ Cx) {
  const int S = 4096, D = 512, HD = 128, BK = 32;
  __shared__ unsigned short Ks[BK][HD];      // 8 KB  [key][d]
  __shared__ unsigned short Vs[BK][HD];      // 8 KB  [key][d]
  __shared__ unsigned short Ps[4][16][BK];   // 4 KB  per-wave P scratch

  const int b = blockIdx.z, h = blockIdx.y;
  const int wave = threadIdx.x >> 5, lane = threadIdx.x & 31;
  const int half = lane >> 4, l16 = lane & 15;
  const int qw = blockIdx.x * 64 + wave * 16;
  const size_t headoff  = (size_t)h * HD;
  const size_t batchrow = (size_t)b * S;

  // Q fragments for this wave's 16 rows: 4 k-tiles over head_dim=128
  BF16x16 Aq[4];
#pragma unroll
  for (int kt = 0; kt < 4; kt++) {
    const unsigned short* p =
        Q + (batchrow + qw + l16) * D + headoff + kt * 32 + half * 16;
    Aq[kt].q[0] = *(const u32x4*)p;
    Aq[kt].q[1] = *(const u32x4*)(p + 8);
  }

  // Online-softmax state. Lanes 0-15 own tile rows r, lanes 16-31 rows r+8.
  float m_r[8], l_r[8];
  F32x8 O[8];
#pragma unroll
  for (int r = 0; r < 8; r++) {
    m_r[r] = -1e30f; l_r[r] = 0.0f;
#pragma unroll
    for (int d = 0; d < 8; d++) O[d].f[r] = 0.0f;
  }

  for (int kb0 = 0; kb0 < S; kb0 += BK) {
    __syncthreads();
    {
      const unsigned short* gk = Kg + (batchrow + kb0) * D + headoff;
      const unsigned short* gv = Vg + (batchrow + kb0) * D + headoff;
#pragma unroll
      for (int it = 0; it < 4; it++) {            // 512 b128s / 128 threads
        int idx = it * 128 + threadIdx.x;
        int f = idx * 8;
        int key = f >> 7, d = f & 127;
#if USE_ASYNC_LDS
        __builtin_amdgcn_global_load_async_to_lds_b128(
            (AS1 i32x4*)(gk + (size_t)key * D + d),
            (AS3 i32x4*)&Ks[key][d], 0, 0);
        __builtin_amdgcn_global_load_async_to_lds_b128(
            (AS1 i32x4*)(gv + (size_t)key * D + d),
            (AS3 i32x4*)&Vs[key][d], 0, 0);
#else
        ((u32x4*)&Ks[0][0])[idx] = *(const u32x4*)(gk + (size_t)key * D + d);
        ((u32x4*)&Vs[0][0])[idx] = *(const u32x4*)(gv + (size_t)key * D + d);
#endif
      }
#if USE_ASYNC_LDS
      __builtin_amdgcn_s_wait_asynccnt(0);
#endif
    }
    __syncthreads();

    // S = Q K^T for 32 keys: 2 n-tiles x 4 k-tiles
    F32x8 Sc[2];
#pragma unroll
    for (int nt = 0; nt < 2; nt++)
#pragma unroll
      for (int r = 0; r < 8; r++) Sc[nt].f[r] = 0.0f;
#pragma unroll
    for (int nt = 0; nt < 2; nt++)
#pragma unroll
      for (int kt = 0; kt < 4; kt++) {
        BF16x16 Bk;  // lane column = key (nt*16+l16); elems = d contiguous
        const unsigned short* p = &Ks[nt * 16 + l16][kt * 32 + half * 16];
        Bk.q[0] = *(const u32x4*)p;
        Bk.q[1] = *(const u32x4*)(p + 8);
        Sc[nt].v = wmma_bf16(Aq[kt].v, Bk.v, Sc[nt].v);
      }

    // Online softmax: row reductions are xor-shuffles across 16 lanes.
    float alpha[8];
#pragma unroll
    for (int r = 0; r < 8; r++) {
      float mx = fmaxf(Sc[0].f[r], Sc[1].f[r]);
      mx = fmaxf(mx, __shfl_xor(mx, 1, 32));
      mx = fmaxf(mx, __shfl_xor(mx, 2, 32));
      mx = fmaxf(mx, __shfl_xor(mx, 4, 32));
      mx = fmaxf(mx, __shfl_xor(mx, 8, 32));
      float mn = fmaxf(m_r[r], mx);
      alpha[r] = __expf(m_r[r] - mn);
      m_r[r] = mn;
      float p0 = __expf(Sc[0].f[r] - mn);
      float p1 = __expf(Sc[1].f[r] - mn);
      Sc[0].f[r] = p0; Sc[1].f[r] = p1;
      float rs = p0 + p1;
      rs += __shfl_xor(rs, 1, 32);
      rs += __shfl_xor(rs, 2, 32);
      rs += __shfl_xor(rs, 4, 32);
      rs += __shfl_xor(rs, 8, 32);
      l_r[r] = l_r[r] * alpha[r] + rs;
    }
#pragma unroll
    for (int dt = 0; dt < 8; dt++)
#pragma unroll
      for (int r = 0; r < 8; r++) O[dt].f[r] *= alpha[r];

    // Re-layout P: C-layout -> LDS row-major -> A-fragment (wave-private).
#pragma unroll
    for (int nt = 0; nt < 2; nt++)
#pragma unroll
      for (int r = 0; r < 8; r++)
        Ps[wave][r + half * 8][nt * 16 + l16] = f2bf(Sc[nt].f[r]);

    BF16x16 Ap;
    {
      const unsigned short* p = &Ps[wave][l16][half * 16];
      Ap.q[0] = *(const u32x4*)p;          // keys kb..kb+7
      Ap.q[1] = *(const u32x4*)(p + 8);    // keys kb+8..kb+15
    }

    // O += P @ V : 8 d-tiles; B lane column = d, elems = key
#pragma unroll
    for (int dt = 0; dt < 8; dt++) {
      BF16x16 Bv;
#pragma unroll
      for (int e = 0; e < 16; e++)
        Bv.s[e] = Vs[half * 16 + e][dt * 16 + l16];
      O[dt].v = wmma_bf16(Ap.v, Bv.v, O[dt].v);
    }
  }

  // Normalize and store context (bf16), layout [B*S, 512] with heads packed.
#pragma unroll
  for (int r = 0; r < 8; r++) {
    const float inv = 1.0f / l_r[r];
    const size_t row = batchrow + qw + r + half * 8;
#pragma unroll
    for (int dt = 0; dt < 8; dt++)
      Cx[row * D + headoff + dt * 16 + l16] = f2bf(O[dt].f[r] * inv);
  }
}

// ---------------------------------------------------------------------------
// Launch
// ---------------------------------------------------------------------------
extern "C" void kernel_launch(void* const* d_in, const int* in_sizes, int n_in,
                              void* d_out, int out_size, void* d_ws,
                              size_t ws_size, hipStream_t stream) {
  const float* q   = (const float*)d_in[0];
  const float* k   = (const float*)d_in[1];
  const float* v   = (const float*)d_in[2];
  const float* Wq  = (const float*)d_in[3];
  const float* bq  = (const float*)d_in[4];
  const float* Wk  = (const float*)d_in[5];
  const float* bk  = (const float*)d_in[6];
  const float* Wv  = (const float*)d_in[7];
  const float* bv  = (const float*)d_in[8];
  const float* Wo  = (const float*)d_in[9];
  const float* bo  = (const float*)d_in[10];

  const size_t WSZ = (size_t)512 * 512;        // one weight matrix (elems)
  const size_t MD  = (size_t)8192 * 512;       // one activation matrix (elems)
  unsigned short* ws  = (unsigned short*)d_ws;
  unsigned short* Wqb = ws;                    // bf16 weights
  unsigned short* Wkb = ws + WSZ;
  unsigned short* Wvb = ws + 2 * WSZ;
  unsigned short* Wob = ws + 3 * WSZ;
  unsigned short* Qb  = ws + 4 * WSZ;          // projected Q/K/V (bf16)
  unsigned short* Kb  = Qb + MD;
  unsigned short* Vb  = Kb + MD;
  unsigned short* Xqb = Vb + MD;               // bf16 activations
  unsigned short* Xkb = Xqb + MD;
  unsigned short* Xvb = Xkb + MD;
  unsigned short* Cx  = Xqb;                   // alias: Xq dead after proj(q)
  // total ws use: 4*WSZ + 6*MD bf16 elems  ~= 52.4 MB

  // 1) fp32 -> bf16 (weights + activations), 4 elems/thread
  const int wn4 = (int)(WSZ / 4);              // 65536
  const int an4 = (int)(MD / 4);               // 1048576
  mha_cvt_bf16x4<<<(wn4 + 255) / 256, 256, 0, stream>>>(Wq, Wqb, wn4);
  mha_cvt_bf16x4<<<(wn4 + 255) / 256, 256, 0, stream>>>(Wk, Wkb, wn4);
  mha_cvt_bf16x4<<<(wn4 + 255) / 256, 256, 0, stream>>>(Wv, Wvb, wn4);
  mha_cvt_bf16x4<<<(wn4 + 255) / 256, 256, 0, stream>>>(Wo, Wob, wn4);
  mha_cvt_bf16x4<<<(an4 + 255) / 256, 256, 0, stream>>>(q, Xqb, an4);
  mha_cvt_bf16x4<<<(an4 + 255) / 256, 256, 0, stream>>>(k, Xkb, an4);
  mha_cvt_bf16x4<<<(an4 + 255) / 256, 256, 0, stream>>>(v, Xvb, an4);

  // 2) projections (2048 waves -> 512 blocks of 128). Scale folded into Q.
  const float scale = 0.08838834764831845f;    // 1/sqrt(128)
  mha_proj_gemm<<<512, 128, 0, stream>>>(Xqb, Wqb, bq, Qb, scale);
  mha_proj_gemm<<<512, 128, 0, stream>>>(Xkb, Wkb, bk, Kb, 1.0f);
  mha_proj_gemm<<<512, 128, 0, stream>>>(Xvb, Wvb, bv, Vb, 1.0f);

  // 3) attention: grid (Sq/64, H, B)   (Cx aliases Xqb, which is now dead)
  mha_attention<<<dim3(64, 4, 2), 128, 0, stream>>>(Qb, Kb, Vb, Cx);

  // 4) output projection -> fp32 d_out
  mha_out_gemm<<<512, 128, 0, stream>>>(Cx, Wob, bo, (float*)d_out);
}
